// InferenceNetwork_78786880078506
// MI455X (gfx1250) — matrix operand order repaired and checked by reading, not compiled, not run on hardware
//
#include <hip/hip_runtime.h>
#include <hip/hip_bf16.h>

// ---------------------------------------------------------------------------
// MI455X / gfx1250: f16 WMMA implicit-GEMM convs (fused relu+maxpool),
// WMMA enc GEMM with async global->LDS staging, scalar tail. wave32.
// ---------------------------------------------------------------------------

typedef _Float16 half16 __attribute__((ext_vector_type(16)));
typedef _Float16 half8  __attribute__((ext_vector_type(8)));
typedef float    float8 __attribute__((ext_vector_type(8)));

#define NIMG   2048
#define WAYS   32
#define XD     84
#define CIN    3
#define C1     10
#define P1     41      // pooled after conv1 (82/2)
#define P2     19      // pooled after conv2 (39/2)
#define FLAT   3610    // 10*19*19
#define FLATP  3616    // padded to multiple of 32
#define ENC    64
#define SX1    86      // padded LDS row stride for conv1 image
#define SX2    44      // padded LDS row stride for conv2 image

#define USE_ASYNC_LDS 1

__device__ __forceinline__ float8 wmma_f16(half16 a, half16 b, float8 c) {
  // 8 args: (neg_a, A, neg_b, B, c_mod, C, reuse_a, reuse_b)
  return __builtin_amdgcn_wmma_f32_16x16x32_f16(false, a, false, b,
                                                (short)0, c, false, false);
}

__device__ __forceinline__ float softplus_f(float x) {
  return (x > 20.f) ? x : log1pf(expf(x));
}

// A-fragment K index for element e (16-bit A 16x32, wave32):
// lane<16: e0..15 -> K0..15 ; lane>=16: K16..31
__device__ __forceinline__ int a_kmap(int e, int half) {
  return e + half * 16;
}

// compile-time patch offsets in the padded LDS image (K -> c,ky,kx)
__device__ constexpr int c1off(int k) {
  return (k / 9) * (XD * SX1) + ((k % 9) / 3) * SX1 + (k % 3);
}
__device__ constexpr int c2off(int k) {
  return (k / 9) * (P1 * SX2) + ((k % 9) / 3) * SX2 + (k % 3);
}

// ---------------------------------------------------------------------------
// Kernel 1: conv1 (3->10, 3x3, VALID) + relu + 2x2 maxpool, f16 output
// one block per image; whole image staged to LDS (padded rows) as f16
// ---------------------------------------------------------------------------
__global__ __launch_bounds__(256)
void conv1_pool_kernel(const float* __restrict__ x,
                       const float* __restrict__ w,
                       const float* __restrict__ bias,
                       _Float16* __restrict__ h1) {
  __shared__ _Float16 sx[CIN * XD * SX1 + 64];   // padded rows + tail pad
  __shared__ _Float16 sw[16 * 32];
  const int n    = blockIdx.x;
  const int tid  = threadIdx.x;
  const int lane = tid & 31;
  const int wave = tid >> 5;
  const int half = lane >> 4;
  const int mrow = lane & 15;

  for (int i = tid; i < CIN * XD * XD; i += 256) {
    int c = i / (XD * XD), rem = i % (XD * XD);
    int r = rem / XD, col = rem % XD;
    sx[c * (XD * SX1) + r * SX1 + col] = (_Float16)x[n * (CIN * XD * XD) + i];
  }
  for (int i = tid; i < 16 * 32; i += 256) sw[i] = (_Float16)0.f;
  __syncthreads();
  for (int i = tid; i < C1 * 27; i += 256) {
    int ch = i / 27, k = i % 27;
    sw[ch * 32 + k] = (_Float16)w[i];
  }
  __syncthreads();

  // A fragment: weights [16 ch x 32 K], constant for whole kernel
  half16 afrag;
#pragma unroll
  for (int e = 0; e < 16; ++e)
    afrag[e] = sw[mrow * 32 + a_kmap(e, half)];

  float8 cbias;
#pragma unroll
  for (int v = 0; v < 8; ++v) {
    int ch = v + half * 8;
    cbias[v] = (ch < C1) ? bias[ch] : 0.f;
  }

  for (int pr = wave; pr < P1; pr += 8) {       // pooled row
    const int r0 = 2 * pr;
    for (int ct = 0; ct < 6; ++ct) {            // 6 x 16 = 96 conv cols (82 used)
      const int cc   = ct * 16 + mrow;          // conv col for this lane
      const int base = r0 * SX1 + cc;
      half16 b0, b1;
#pragma unroll
      for (int e = 0; e < 16; ++e) {
        // B layout: lane<16 -> K=e ; lane>=16 -> K=16+e
        const int oA = c1off(e);
        const int oB = c1off((e + 16 < 27) ? e + 16 : 0);
        const int off = base + (half ? oB : oA);
        _Float16 v0 = sx[off];
        _Float16 v1 = sx[off + SX1];
        const bool ok = (half == 0) || (e < 11);    // K<27
        b0[e] = ok ? v0 : (_Float16)0.f;
        b1[e] = ok ? v1 : (_Float16)0.f;
      }
      float8 d0 = wmma_f16(afrag, b0, cbias);
      float8 d1 = wmma_f16(afrag, b1, cbias);
      const int pc = ct * 8 + (mrow >> 1);
      const bool st = ((mrow & 1) == 0) && (pc < P1);
#pragma unroll
      for (int v = 0; v < 8; ++v) {
        float p = fmaxf(d0[v], d1[v]);                 // row max
        float q = fmaxf(p, __shfl_xor(p, 1, 32));      // col-pair max
        q = fmaxf(q, 0.f);                             // relu (monotone w/ pool)
        int ch = v + half * 8;
        if (st && ch < C1)
          h1[((n * C1 + ch) * P1 + pr) * P1 + pc] = (_Float16)q;
      }
    }
  }
}

// ---------------------------------------------------------------------------
// Kernel 2: conv2 (10->10, 3x3) + relu + 2x2 maxpool, flat f16 output padded
// ---------------------------------------------------------------------------
__global__ __launch_bounds__(256)
void conv2_pool_kernel(const _Float16* __restrict__ h1,
                       const float* __restrict__ w,
                       const float* __restrict__ bias,
                       _Float16* __restrict__ h2) {
  __shared__ _Float16 sh[C1 * P1 * SX2 + 64];
  __shared__ _Float16 sw[16 * 96];        // K = 90 padded to 96
  const int n    = blockIdx.x;
  const int tid  = threadIdx.x;
  const int lane = tid & 31;
  const int wave = tid >> 5;
  const int half = lane >> 4;
  const int mrow = lane & 15;

  for (int i = tid; i < C1 * P1 * P1; i += 256) {
    int c = i / (P1 * P1), rem = i % (P1 * P1);
    int r = rem / P1, col = rem % P1;
    sh[c * (P1 * SX2) + r * SX2 + col] = h1[n * (C1 * P1 * P1) + i];
  }
  for (int i = tid; i < 16 * 96; i += 256) sw[i] = (_Float16)0.f;
  __syncthreads();
  for (int i = tid; i < C1 * 90; i += 256) {
    int ch = i / 90, k = i % 90;
    sw[ch * 96 + k] = (_Float16)w[i];
  }
  if (tid < FLATP - FLAT) h2[n * FLATP + FLAT + tid] = (_Float16)0.f;
  __syncthreads();

  half16 af[3];
#pragma unroll
  for (int s = 0; s < 3; ++s)
#pragma unroll
    for (int e = 0; e < 16; ++e)
      af[s][e] = sw[mrow * 96 + s * 32 + a_kmap(e, half)];

  float8 cbias;
#pragma unroll
  for (int v = 0; v < 8; ++v) {
    int ch = v + half * 8;
    cbias[v] = (ch < C1) ? bias[ch] : 0.f;
  }

  for (int pr = wave; pr < P2; pr += 8) {
    const int r0 = 2 * pr;
    for (int ct = 0; ct < 3; ++ct) {          // 3 x 16 = 48 conv cols (39 used)
      const int cc   = ct * 16 + mrow;
      const int base = r0 * SX2 + cc;
      float8 d0 = cbias, d1 = cbias;
#pragma unroll
      for (int s = 0; s < 3; ++s) {
        half16 b0, b1;
#pragma unroll
        for (int e = 0; e < 16; ++e) {
          const int k0 = s * 32 + e;          // lane<16   (always < 90)
          const int k1 = s * 32 + 16 + e;     // lane>=16  (invalid iff s==2, e>=10)
          const int oA = c2off(k0);
          const int oB = c2off((k1 < 90) ? k1 : 0);
          const int off = base + (half ? oB : oA);
          _Float16 v0 = sh[off];
          _Float16 v1 = sh[off + SX2];
          const bool ok = (half == 0) || (k1 < 90);
          b0[e] = ok ? v0 : (_Float16)0.f;
          b1[e] = ok ? v1 : (_Float16)0.f;
        }
        d0 = wmma_f16(af[s], b0, d0);
        d1 = wmma_f16(af[s], b1, d1);
      }
      const int pc = ct * 8 + (mrow >> 1);
      const bool st = ((mrow & 1) == 0) && (pc < P2);
#pragma unroll
      for (int v = 0; v < 8; ++v) {
        float p = fmaxf(d0[v], d1[v]);
        float q = fmaxf(p, __shfl_xor(p, 1, 32));
        q = fmaxf(q, 0.f);
        int ch = v + half * 8;
        if (st && ch < C1)
          h2[n * FLATP + ch * (P2 * P2) + pr * P2 + pc] = (_Float16)q;
      }
    }
  }
}

// ---------------------------------------------------------------------------
// Kernel 3a: convert enc_w [3610x64] f32 -> [3616x64] f16 (zero-padded rows)
// ---------------------------------------------------------------------------
__global__ __launch_bounds__(256)
void encw_cvt_kernel(const float* __restrict__ enc_w, _Float16* __restrict__ w16) {
  int idx = blockIdx.x * 256 + threadIdx.x;
  if (idx < FLATP * ENC) {
    int k = idx >> 6, c = idx & 63;
    w16[idx] = (k < FLAT) ? (_Float16)enc_w[k * ENC + c] : (_Float16)0.f;
  }
}

// ---------------------------------------------------------------------------
// Kernel 3b: enc GEMM henc[2048x64] = h2[2048x3616] @ w16[3616x64] + b
// block = 64 M-rows, full N=64; 8 waves x 2 (16x16) tiles
// A tile staged via CDNA5 async global->LDS (b128, one per thread)
// ---------------------------------------------------------------------------
__global__ __launch_bounds__(256)
void enc_gemm_kernel(const _Float16* __restrict__ h2,
                     const _Float16* __restrict__ w16,
                     const float* __restrict__ bias,
                     float* __restrict__ henc) {
  __shared__ __align__(16) _Float16 sA[64 * 32];    // A tile  [64 M][32 K]
  __shared__ __align__(32) _Float16 sBt[64 * 32];   // B tile transposed [64 N][32 K]
  const int m0   = blockIdx.x * 64;
  const int tid  = threadIdx.x;
  const int lane = tid & 31;
  const int wave = tid >> 5;
  const int half = lane >> 4;
  const int mrow = lane & 15;

  float8 acc[2] = {{}, {}};
  const int tiles[2] = {wave, wave + 8};   // of 16 tiles (4 M-tiles x 4 N-tiles)

#if USE_ASYNC_LDS
  // low 32 bits of a flat shared address == LDS byte offset
  const unsigned ldsA = (unsigned)(size_t)(&sA[0]) + (unsigned)tid * 16u;
#endif

  for (int kk = 0; kk < FLATP / 32; ++kk) {
#if USE_ASYNC_LDS
    {
      // 64x32 f16 tile = 4096 B = 256 threads x 16 B
      const _Float16* g =
          &h2[(size_t)(m0 + (tid >> 2)) * FLATP + kk * 32 + (tid & 3) * 8];
      asm volatile("global_load_async_to_lds_b128 %0, %1, off"
                   :: "v"(ldsA), "v"(g) : "memory");
    }
#else
    for (int i = tid; i < 64 * 32; i += 256) {
      int r = i >> 5, k = i & 31;
      sA[r * 32 + k] = h2[(size_t)(m0 + r) * FLATP + kk * 32 + k];
    }
#endif
    if (kk + 1 < FLATP / 32)
      __builtin_prefetch(&w16[(kk + 1) * 32 * ENC + tid * 8], 0, 1);
    for (int i = tid; i < 32 * 64; i += 256) {
      int k = i >> 6, c = i & 63;                      // coalesced global read
      sBt[c * 32 + k] = w16[(kk * 32 + k) * ENC + c];  // transposed LDS write
    }
#if USE_ASYNC_LDS
    asm volatile("s_wait_asynccnt 0x0" ::: "memory");
#endif
    __syncthreads();
#pragma unroll
    for (int ti = 0; ti < 2; ++ti) {
      const int t  = tiles[ti];
      const int tm = t >> 2, tn = t & 3;
      const int M  = tm * 16 + mrow;
      half8 lo = *(const half8*)&sA[M * 32 + half * 16];
      half8 hi = *(const half8*)&sA[M * 32 + half * 16 + 8];
      half16 a = __builtin_shufflevector(lo, hi, 0, 1, 2, 3, 4, 5, 6, 7,
                                                  8, 9, 10, 11, 12, 13, 14, 15);
      const int Nc = tn * 16 + mrow;
      half16 b = *(const half16*)&sBt[Nc * 32 + half * 16];
      acc[ti] = wmma_f16(a, b, acc[ti]);
    }
    __syncthreads();
  }
#pragma unroll
  for (int ti = 0; ti < 2; ++ti) {
    const int t  = tiles[ti];
    const int tm = t >> 2, tn = t & 3;
    const int Nc = tn * 16 + mrow;
#pragma unroll
    for (int v = 0; v < 8; ++v) {
      int M = tm * 16 + v + half * 8;
      henc[(size_t)(m0 + M) * ENC + Nc] = acc[ti][v] + bias[Nc];
    }
  }
}

// ---------------------------------------------------------------------------
// Kernel 4: per-class masked mean / unbiased var / cardinality + t1 head
// ---------------------------------------------------------------------------
__global__ __launch_bounds__(256)
void stats1_kernel(const float* __restrict__ henc, const float* __restrict__ y,
                   const float* __restrict__ t1w, const float* __restrict__ t1b,
                   float* __restrict__ cs, float* __restrict__ card) {
  __shared__ float sS[4][64], sQ[4][64], sC[4];
  const int w = blockIdx.x, tid = threadIdx.x;
  const int g = tid >> 6, f = tid & 63;
  float S = 0.f, Q = 0.f, cnt = 0.f;
  for (int i = g * 512; i < (g + 1) * 512; ++i) {
    if (y[i * WAYS + w] > 0.5f) {                 // one-hot -> argmax==w
      float v = henc[i * ENC + f];
      S += v; Q += v * v; cnt += 1.f;
    }
  }
  sS[g][f] = S; sQ[g][f] = Q;
  if (f == 0) sC[g] = cnt;
  __syncthreads();
  if (tid < 64) {
    float St = 0.f, Qt = 0.f;
    for (int gg = 0; gg < 4; ++gg) { St += sS[gg][tid]; Qt += sQ[gg][tid]; }
    float mean = St / 2048.f;
    float var  = (Qt - St * St / 2048.f) / 2047.f;   // ddof=1 over masked-zeros
    float cn   = sC[0] + sC[1] + sC[2] + sC[3];
    float cd   = (cn - 1.f) / 63.f;
#pragma unroll
    for (int j = 0; j < 4; ++j) {
      float v = mean * t1w[j] + var * t1w[4 + j] + cd * t1w[8 + j] + t1b[j];
      cs[w * 256 + tid * 4 + j] = fmaxf(v, 0.f);
    }
    if (tid == 0) card[w] = cd;
  }
}

// ---------------------------------------------------------------------------
// Kernel 5: tail — te2 MLP, stats_pool_2, t2, omega/gamma/zeta heads, KL
// ---------------------------------------------------------------------------
__global__ __launch_bounds__(256)
void tail_kernel(const float* __restrict__ cs, const float* __restrict__ card,
                 const float* te2w1, const float* te2b1,
                 const float* te2w2, const float* te2b2,
                 const float* t2w, const float* t2b,
                 const float* omw, const float* omb,
                 const float* osw, const float* osb,
                 const float* gaw, const float* gab,
                 const float* gmw, const float* gmb,
                 const float* gsw, const float* gsb,
                 const float* zew, const float* zeb,
                 const float* zmw, const float* zmb,
                 const float* zsw, const float* zsb,
                 float* __restrict__ out) {
  __shared__ float sH[32 * 128];
  __shared__ float sTs[32 * 32];
  __shared__ float sT[128];
  __shared__ float sEg[64], sEz[64];
  __shared__ float sKL, sCard;
  const int tid = threadIdx.x;
  if (tid == 0) {
    sKL = 0.f;
    float c = 0.f;
    for (int i = 0; i < WAYS; ++i) c += card[i];
    sCard = c;
  }
  __syncthreads();
  // te2 layer 1: relu(cs[32x256] @ w1[256x128] + b1)
  for (int idx = tid; idx < 32 * 128; idx += 256) {
    int c = idx >> 7, j = idx & 127;
    float a = te2b1[j];
    for (int k = 0; k < 256; ++k) a += cs[c * 256 + k] * te2w1[k * 128 + j];
    sH[idx] = fmaxf(a, 0.f);
  }
  __syncthreads();
  // te2 layer 2 (no relu)
  for (int idx = tid; idx < 32 * 32; idx += 256) {
    int c = idx >> 5, j = idx & 31;
    float a = te2b2[j];
    for (int k = 0; k < 128; ++k) a += sH[c * 128 + k] * te2w2[k * 32 + j];
    sTs[idx] = a;
  }
  __syncthreads();
  // stats over classes + t2
  if (tid < 32) {
    float S = 0.f, Q = 0.f;
    for (int c = 0; c < 32; ++c) { float v = sTs[c * 32 + tid]; S += v; Q += v * v; }
    float mean = S / 32.f;
    float var  = (Q - S * S / 32.f) / 31.f;
#pragma unroll
    for (int q = 0; q < 4; ++q) {
      float v = mean * t2w[q] + var * t2w[4 + q] + sCard * t2w[8 + q] + t2b[q];
      sT[tid * 4 + q] = fmaxf(v, 0.f);
    }
  }
  __syncthreads();
  // omega heads (applied to class summaries directly)
  if (tid < 32) {
    float m = omb[0], s = osb[0];
    for (int k = 0; k < 256; ++k) {
      float c = cs[tid * 256 + k];
      m += c * omw[k]; s += c * osw[k];
    }
    float mu = fmaxf(m, 0.f);
    float sd = softplus_f(fmaxf(s, 0.f));
    out[tid] = mu;
    atomicAdd(&sKL, -logf(sd) + 0.5f * (sd * sd + mu * mu) - 0.5f);
  }
  // gamma / zeta encoders
  if (tid < 64) {
    float a = gab[tid], z = zeb[tid];
    for (int k = 0; k < 128; ++k) {
      float t = sT[k];
      a += t * gaw[k * 64 + tid];
      z += t * zew[k * 64 + tid];
    }
    sEg[tid] = fmaxf(a, 0.f);
    sEz[tid] = fmaxf(z, 0.f);
  }
  __syncthreads();
  if (tid < 5) {
    float m = gmb[tid], s = gsb[tid];
    for (int k = 0; k < 64; ++k) { float e = sEg[k]; m += e * gmw[k * 5 + tid]; s += e * gsw[k * 5 + tid]; }
    float mu = fmaxf(m, 0.f);
    float sd = softplus_f(fmaxf(s, 0.f));
    out[32 + tid] = mu;
    atomicAdd(&sKL, -logf(sd) + 0.5f * (sd * sd + mu * mu) - 0.5f);
  }
  {
    float m = zmb[tid], s = zsb[tid];
    for (int k = 0; k < 64; ++k) { float e = sEz[k]; m += e * zmw[k * 256 + tid]; s += e * zsw[k * 256 + tid]; }
    float mu = fmaxf(m, 0.f);
    float sd = softplus_f(fmaxf(s, 0.f));
    out[37 + tid] = mu;
    atomicAdd(&sKL, -logf(sd) + 0.5f * (sd * sd + mu * mu) - 0.5f);
  }
  __syncthreads();
  if (tid == 0) out[37 + 256] = sKL;   // out[293]
}

// ---------------------------------------------------------------------------
// launcher
// ---------------------------------------------------------------------------
static inline size_t align256(size_t x) { return (x + 255) & ~(size_t)255; }

extern "C" void kernel_launch(void* const* d_in, const int* in_sizes, int n_in,
                              void* d_out, int out_size, void* d_ws, size_t ws_size,
                              hipStream_t stream) {
  (void)in_sizes; (void)n_in; (void)out_size; (void)ws_size;
  const float* x    = (const float*)d_in[0];
  const float* y    = (const float*)d_in[1];
  const float* c1w  = (const float*)d_in[2];
  const float* c1b  = (const float*)d_in[3];
  const float* c2w  = (const float*)d_in[4];
  const float* c2b  = (const float*)d_in[5];
  const float* encw = (const float*)d_in[6];
  const float* encb = (const float*)d_in[7];
  const float* t1w  = (const float*)d_in[8];
  const float* t1b  = (const float*)d_in[9];
  const float* te2w1 = (const float*)d_in[10];
  const float* te2b1 = (const float*)d_in[11];
  const float* te2w2 = (const float*)d_in[12];
  const float* te2b2 = (const float*)d_in[13];
  const float* t2w  = (const float*)d_in[14];
  const float* t2b  = (const float*)d_in[15];
  const float* omw  = (const float*)d_in[16];
  const float* omb  = (const float*)d_in[17];
  const float* osw  = (const float*)d_in[18];
  const float* osb  = (const float*)d_in[19];
  const float* gaw  = (const float*)d_in[20];
  const float* gab  = (const float*)d_in[21];
  const float* gmw  = (const float*)d_in[22];
  const float* gmb  = (const float*)d_in[23];
  const float* gsw  = (const float*)d_in[24];
  const float* gsb  = (const float*)d_in[25];
  const float* zew  = (const float*)d_in[26];
  const float* zeb  = (const float*)d_in[27];
  const float* zmw  = (const float*)d_in[28];
  const float* zmb  = (const float*)d_in[29];
  const float* zsw  = (const float*)d_in[30];
  const float* zsb  = (const float*)d_in[31];
  float* out = (float*)d_out;

  // workspace layout
  char* ws = (char*)d_ws;
  size_t o = 0;
  _Float16* h1 = (_Float16*)(ws + o); o = align256(o + (size_t)NIMG * C1 * P1 * P1 * 2);
  _Float16* h2 = (_Float16*)(ws + o); o = align256(o + (size_t)NIMG * FLATP * 2);
  _Float16* w16 = (_Float16*)(ws + o); o = align256(o + (size_t)FLATP * ENC * 2);
  float* henc = (float*)(ws + o); o = align256(o + (size_t)NIMG * ENC * 4);
  float* cs   = (float*)(ws + o); o = align256(o + (size_t)WAYS * 256 * 4);
  float* card = (float*)(ws + o); o = align256(o + (size_t)WAYS * 4);

  conv1_pool_kernel<<<dim3(NIMG), dim3(256), 0, stream>>>(x, c1w, c1b, h1);
  conv2_pool_kernel<<<dim3(NIMG), dim3(256), 0, stream>>>(h1, c2w, c2b, h2);
  encw_cvt_kernel<<<dim3((FLATP * ENC + 255) / 256), dim3(256), 0, stream>>>(encw, w16);
  enc_gemm_kernel<<<dim3(NIMG / 64), dim3(256), 0, stream>>>(h2, w16, encb, henc);
  stats1_kernel<<<dim3(WAYS), dim3(256), 0, stream>>>(henc, y, t1w, t1b, cs, card);
  tail_kernel<<<dim3(1), dim3(256), 0, stream>>>(cs, card,
      te2w1, te2b1, te2w2, te2b2, t2w, t2b,
      omw, omb, osw, osb, gaw, gab, gmw, gmb, gsw, gsb,
      zew, zeb, zmw, zmb, zsw, zsb, out);
}
